// RGATStack_77283641524510
// MI455X (gfx1250) — matrix operand order repaired and checked by reading, not compiled, not run on hardware
//
#include <hip/hip_runtime.h>

// Problem constants (from reference)
#define NN   20000
#define EE   320000
#define CIN_ 256
#define CC   256
#define LL   2
#define HH   8
#define HD_  32
#define RR   64
#define FFN_ 1024

#define BM 128
#define BN 128
#define BK 32
#define LDA 40   // padded LDS stride (elements) to decorrelate banks

typedef __attribute__((ext_vector_type(16))) __bf16 v16bf;
typedef __attribute__((ext_vector_type(8)))  __bf16 v8bf;
typedef __attribute__((ext_vector_type(8)))  float  v8f;

__device__ __forceinline__ unsigned short f2bf(float f) {
  unsigned int u = __float_as_uint(f);
  unsigned int r = (u + 0x7FFFu + ((u >> 16) & 1u)) >> 16;  // RNE
  return (unsigned short)r;
}

__device__ __forceinline__ float gelu_exact(float x) {
  return 0.5f * x * (1.0f + erff(x * 0.7071067811865475f));
}

__device__ __forceinline__ void atomicMaxF(float* addr, float value) {
  if (value >= 0.0f) atomicMax((int*)addr, __float_as_int(value));
  else               atomicMin((unsigned int*)addr, __float_as_uint(value));
}

// CDNA5 async global->LDS copy (16B per enabled lane), tracked by ASYNCcnt.
// ldsoff: wave-relative LDS byte address (generic pointer low 32 bits).
__device__ __forceinline__ void async_copy_b128(unsigned ldsoff,
                                                unsigned long long gaddr) {
  asm volatile("global_load_async_to_lds_b128 %0, %1, off"
               :: "v"(ldsoff), "v"(gaddr)
               : "memory");
}

__device__ __forceinline__ void wait_async0() {
#if __has_builtin(__builtin_amdgcn_s_wait_asynccnt)
  __builtin_amdgcn_s_wait_asynccnt(0);
#else
  asm volatile("s_wait_asynccnt 0" ::: "memory");
#endif
}

// ---------------- elementwise helpers ----------------

__global__ void cvt_f32_bf16(const float* __restrict__ src,
                             unsigned short* __restrict__ dst, int n) {
  int i = blockIdx.x * 256 + threadIdx.x;
  if (i < n) dst[i] = f2bf(src[i]);
}

__global__ void zero_f32(float* __restrict__ p, int n) {
  int i = blockIdx.x * 256 + threadIdx.x;
  if (i < n) p[i] = 0.0f;
}

__global__ void init_mz(float* __restrict__ m, float* __restrict__ z, int n) {
  int i = blockIdx.x * 256 + threadIdx.x;
  if (i < n) { ((unsigned int*)m)[i] = 0xFF800000u; z[i] = 0.0f; }  // -inf, 0
}

// Transpose-convert weight: W[K][Nc] f32 -> Wt[Nc][K] bf16 (coalesced writes)
__global__ void wtrans(const float* __restrict__ W,
                       unsigned short* __restrict__ Wt, int K, int Nc) {
  int i = blockIdx.x * 256 + threadIdx.x;
  if (i >= K * Nc) return;
  int n = i / K, k = i - n * K;
  Wt[i] = f2bf(W[(size_t)k * Nc + n]);
}

// RMSNorm row kernel: out_bf16[row][c] = x*rsqrt(mean(x^2)+EPS)*w[c]; C==256
__global__ void rmsnorm_kernel(const float* __restrict__ x,
                               const float* __restrict__ w,
                               unsigned short* __restrict__ out, int rows) {
  int row = blockIdx.x;
  int t = threadIdx.x;
  if (row >= rows) return;
  float v = x[(size_t)row * CC + t];
  float sq = v * v;
  #pragma unroll
  for (int off = 16; off; off >>= 1) sq += __shfl_xor(sq, off, 32);
  __shared__ float red[8];
  __shared__ float stot;
  if ((t & 31) == 0) red[t >> 5] = sq;
  __syncthreads();
  if (t == 0) {
    float s = 0.f;
    #pragma unroll
    for (int i = 0; i < 8; i++) s += red[i];
    stot = s * (1.0f / (float)CC);
  }
  __syncthreads();
  float scale = rsqrtf(stot + 1.1920929e-07f);
  out[(size_t)row * CC + t] = f2bf(v * scale * w[t]);
}

// ---------------- WMMA GEMM (async-LDS staged, double buffered) ----------------
// out[M][Nc] = A_bf16[M][K] @ Bt_bf16[Nc][K]^T + bias, with epilogue MODE:
//   0: f32 out = acc+bias
//   1: f32 out = acc+bias+res
//   2: bf16 out = gelu(acc+bias)
// 256 threads = 8 waves (wave32), 4x2 wave grid, each wave 32x64 (2x4 WMMA).
// Tiles staged with global_load_async_to_lds_b128; next tile's loads overlap
// the current tile's WMMAs (one s_wait_asynccnt + one barrier per K-step).
template <int MODE>
__global__ __launch_bounds__(256) void gemm_bf16(
    const unsigned short* __restrict__ A, const unsigned short* __restrict__ Bt,
    const float* __restrict__ bias, const float* __restrict__ res,
    void* __restrict__ outp, int M, int Nc, int K) {
  __shared__ __align__(16) unsigned short ldsA[2 * BM * LDA];
  __shared__ __align__(16) unsigned short ldsB[2 * BN * LDA];

  const int t = threadIdx.x;
  const int lane = t & 31;
  const int wv = t >> 5;
  const int wr = wv & 3;   // wave row   (4 -> 128 rows)
  const int wc = wv >> 2;  // wave col   (2 -> 128 cols)
  const int tileM = blockIdx.x * BM;
  const int tileN = blockIdx.y * BN;
  const int lh = lane >> 4;  // lane half
  const int lm = lane & 15;

  v8f acc[2][4];
  #pragma unroll
  for (int i = 0; i < 2; i++)
    #pragma unroll
    for (int j = 0; j < 4; j++)
      #pragma unroll
      for (int r = 0; r < 8; r++) acc[i][j][r] = 0.0f;

  // Per-thread staging slice: row = t/2, 16 elements (32B) at kpos = (t&1)*16.
  const int lrow = t >> 1;
  const int lkp = (t & 1) * 16;
  const int grow = tileM + lrow;
  const unsigned bufBytes = (unsigned)(BM * LDA * 2);
  const unsigned ldsAoff =
      (unsigned)(unsigned long long)(void*)&ldsA[lrow * LDA + lkp];
  const unsigned ldsBoff =
      (unsigned)(unsigned long long)(void*)&ldsB[lrow * LDA + lkp];
  const unsigned long long gA =
      (unsigned long long)(A + (size_t)grow * K + lkp);
  const unsigned long long gB =
      (unsigned long long)(Bt + (size_t)(tileN + lrow) * K + lkp);

  auto stage = [&](int k0, int buf) {
    unsigned la = ldsAoff + (unsigned)buf * bufBytes;
    unsigned lb = ldsBoff + (unsigned)buf * bufBytes;
    unsigned long long ga = gA + (unsigned long long)k0 * 2ull;
    unsigned long long gb = gB + (unsigned long long)k0 * 2ull;
    if (grow < M) {  // tail rows: skip (garbage only feeds unstored D rows)
      async_copy_b128(la, ga);
      async_copy_b128(la + 16u, ga + 16ull);
    }
    async_copy_b128(lb, gb);
    async_copy_b128(lb + 16u, gb + 16ull);
  };

  const int nk = K / BK;
  stage(0, 0);

  for (int ik = 0; ik < nk; ik++) {
    const int buf = ik & 1;
    wait_async0();       // this wave's slice of tile `buf` is in LDS
    __syncthreads();     // whole tile visible; prev readers of buf^1 are done
    if (ik + 1 < nk) stage((ik + 1) * BK, buf ^ 1);

    const unsigned short* sA = ldsA + buf * (BM * LDA);
    const unsigned short* sB = ldsB + buf * (BN * LDA);

    union Frag { v16bf v; v8bf h[2]; };
    Frag fa[2], fb[4];
    // A frag (16x32 bf16): lanes 0-15 row=lm K{0-7,16-23}; lanes 16-31 K{8-15,24-31}
    #pragma unroll
    for (int i = 0; i < 2; i++) {
      const unsigned short* p = &sA[(wr * 32 + i * 16 + lm) * LDA + lh * 8];
      fa[i].h[0] = *(const v8bf*)(p);
      fa[i].h[1] = *(const v8bf*)(p + 16);
    }
    // B frag (32x16 bf16): lane holds column N=lm; lanes 0-15 K{0-7,16-23}; lanes 16-31 K{8-15,24-31}
    #pragma unroll
    for (int j = 0; j < 4; j++) {
      const unsigned short* p = &sB[(wc * 64 + j * 16 + lm) * LDA + lh * 8];
      fb[j].h[0] = *(const v8bf*)(p);
      fb[j].h[1] = *(const v8bf*)(p + 16);
    }
    #pragma unroll
    for (int i = 0; i < 2; i++)
      #pragma unroll
      for (int j = 0; j < 4; j++)
        acc[i][j] = __builtin_amdgcn_wmma_f32_16x16x32_bf16(
            false, fa[i].v, false, fb[j].v, (short)0, acc[i][j], false, false);
  }

  // Epilogue: C/D layout — VGPR r: lanes0-15 M=r, lanes16-31 M=8+r, N=lm
  #pragma unroll
  for (int i = 0; i < 2; i++) {
    #pragma unroll
    for (int j = 0; j < 4; j++) {
      int colg = tileN + wc * 64 + j * 16 + lm;
      float bb = bias[colg];
      #pragma unroll
      for (int r = 0; r < 8; r++) {
        int rowg = tileM + wr * 32 + i * 16 + lh * 8 + r;
        if (rowg < M) {
          float v = acc[i][j][r] + bb;
          size_t o = (size_t)rowg * Nc + colg;
          if (MODE == 0)      ((float*)outp)[o] = v;
          else if (MODE == 1) ((float*)outp)[o] = v + res[o];
          else                ((unsigned short*)outp)[o] = f2bf(gelu_exact(v));
        }
      }
    }
  }
}

// ---------------- edge-phase kernels (wave per edge, lane = head-dim) ------

__global__ void edge_score(const float* __restrict__ q, const float* __restrict__ k,
                           const float* __restrict__ rel_l,
                           const int* __restrict__ src, const int* __restrict__ dst,
                           const int* __restrict__ etype,
                           float* __restrict__ score, float* __restrict__ m, int E_) {
  int wid = (blockIdx.x * blockDim.x + threadIdx.x) >> 5;
  int lane = threadIdx.x & 31;
  if (wid >= E_) return;
  int s = src[wid], d = dst[wid], r = etype[wid];
  const float* qp = q + (size_t)d * CC;
  const float* kp = k + (size_t)s * CC;
  const float* rp = rel_l + (size_t)r * CC;
  #pragma unroll
  for (int h = 0; h < HH; h++) {
    int idx = h * HD_ + lane;
    float v = qp[idx] * (kp[idx] + rp[idx]);
    #pragma unroll
    for (int off = 16; off; off >>= 1) v += __shfl_xor(v, off, 32);
    if (lane == 0) {
      float sc = v * 0.17677669529663687f;  // 1/sqrt(32)
      score[(size_t)wid * HH + h] = sc;
      atomicMaxF(&m[(size_t)d * HH + h], sc);
    }
  }
}

__global__ void edge_expsum(float* __restrict__ score, const float* __restrict__ m,
                            float* __restrict__ z, const int* __restrict__ dst, int E_) {
  int t = blockIdx.x * 256 + threadIdx.x;
  if (t >= E_ * HH) return;
  int e = t >> 3, h = t & 7;
  int d = dst[e];
  float ex = expf(score[t] - m[(size_t)d * HH + h]);
  score[t] = ex;
  atomicAdd(&z[(size_t)d * HH + h], ex);
}

__global__ void edge_agg(const float* __restrict__ vbuf, const float* __restrict__ rel_l,
                         const int* __restrict__ src, const int* __restrict__ dst,
                         const int* __restrict__ etype,
                         const float* __restrict__ score, const float* __restrict__ z,
                         float* __restrict__ agg, int E_) {
  int wid = (blockIdx.x * blockDim.x + threadIdx.x) >> 5;
  int lane = threadIdx.x & 31;
  if (wid >= E_) return;
  int s = src[wid], d = dst[wid], r = etype[wid];
  const float* vp = vbuf + (size_t)s * CC;
  const float* rp = rel_l + (size_t)r * CC;
  float* ap = agg + (size_t)d * CC;
  #pragma unroll
  for (int h = 0; h < HH; h++) {
    float p = score[(size_t)wid * HH + h] / (z[(size_t)d * HH + h] + 1e-16f);
    int idx = h * HD_ + lane;
    atomicAdd(&ap[idx], p * (vp[idx] + rp[idx]));
  }
}

// ---------------- host orchestration ----------------

extern "C" void kernel_launch(void* const* d_in, const int* in_sizes, int n_in,
                              void* d_out, int out_size, void* d_ws, size_t ws_size,
                              hipStream_t stream) {
  (void)in_sizes; (void)n_in; (void)out_size; (void)ws_size;
  const float* x_in  = (const float*)d_in[0];
  const int*   eidx  = (const int*)d_in[1];
  const int*   etype = (const int*)d_in[2];
  const float* Wp = (const float*)d_in[3];
  const float* bp = (const float*)d_in[4];
  const float* Wq = (const float*)d_in[5];
  const float* bq = (const float*)d_in[6];
  const float* Wk = (const float*)d_in[7];
  const float* bk = (const float*)d_in[8];
  const float* Wv = (const float*)d_in[9];
  const float* bv = (const float*)d_in[10];
  const float* rel = (const float*)d_in[11];
  const float* Wo = (const float*)d_in[12];
  const float* bo = (const float*)d_in[13];
  const float* n1w = (const float*)d_in[14];
  const float* n2w = (const float*)d_in[15];
  const float* W1 = (const float*)d_in[16];
  const float* b1 = (const float*)d_in[17];
  const float* W2 = (const float*)d_in[18];
  const float* b2 = (const float*)d_in[19];
  const int* srcv = eidx;
  const int* dstv = eidx + EE;

  char* base = (char*)d_ws;
  size_t off = 0;
  auto alloc = [&](size_t bytes) -> void* {
    void* p = base + off;
    off = (off + bytes + 255) & ~(size_t)255;
    return p;
  };

  float* xcur = (float*)alloc((size_t)NN * CC * 4);
  unsigned short* xbf = (unsigned short*)alloc((size_t)NN * CC * 2);
  float* qb  = (float*)alloc((size_t)NN * CC * 4);
  float* kb  = (float*)alloc((size_t)NN * CC * 4);
  float* vb  = (float*)alloc((size_t)NN * CC * 4);
  float* agg = (float*)alloc((size_t)NN * CC * 4);
  float* score = (float*)alloc((size_t)EE * HH * 4);
  float* mb = (float*)alloc((size_t)NN * HH * 4);
  float* zb = (float*)alloc((size_t)NN * HH * 4);
  unsigned short* hbuf = (unsigned short*)alloc((size_t)NN * FFN_ * 2);
  unsigned short* WpT = (unsigned short*)alloc((size_t)CIN_ * CC * 2);
  unsigned short* WqT = (unsigned short*)alloc((size_t)LL * CC * CC * 2);
  unsigned short* WkT = (unsigned short*)alloc((size_t)LL * CC * CC * 2);
  unsigned short* WvT = (unsigned short*)alloc((size_t)LL * CC * CC * 2);
  unsigned short* WoT = (unsigned short*)alloc((size_t)LL * CC * CC * 2);
  unsigned short* W1T = (unsigned short*)alloc((size_t)LL * CC * FFN_ * 2);
  unsigned short* W2T = (unsigned short*)alloc((size_t)LL * CC * FFN_ * 2);

  auto launchW = [&](const float* W, unsigned short* Wt, int K, int Nc) {
    int n = K * Nc;
    wtrans<<<(n + 255) / 256, 256, 0, stream>>>(W, Wt, K, Nc);
  };

  // Weight prep (bf16, transposed [N][K] so the GEMM streams K contiguously)
  launchW(Wp, WpT, CIN_, CC);
  for (int l = 0; l < LL; l++) {
    launchW(Wq + (size_t)l * CC * CC, WqT + (size_t)l * CC * CC, CC, CC);
    launchW(Wk + (size_t)l * CC * CC, WkT + (size_t)l * CC * CC, CC, CC);
    launchW(Wv + (size_t)l * CC * CC, WvT + (size_t)l * CC * CC, CC, CC);
    launchW(Wo + (size_t)l * CC * CC, WoT + (size_t)l * CC * CC, CC, CC);
    launchW(W1 + (size_t)l * CC * FFN_, W1T + (size_t)l * CC * FFN_, CC, FFN_);
    launchW(W2 + (size_t)l * FFN_ * CC, W2T + (size_t)l * FFN_ * CC, FFN_, CC);
  }

  dim3 gC((NN + BM - 1) / BM, CC / BN);
  dim3 gF((NN + BM - 1) / BM, FFN_ / BN);

  // input projection: xcur = x @ Wp + bp
  cvt_f32_bf16<<<((NN * CIN_) + 255) / 256, 256, 0, stream>>>(x_in, xbf, NN * CIN_);
  gemm_bf16<0><<<gC, 256, 0, stream>>>(xbf, WpT, bp, nullptr, xcur, NN, CC, CIN_);

  for (int l = 0; l < LL; l++) {
    const float* rel_l = rel + (size_t)l * RR * CC;
    // attention sublayer
    rmsnorm_kernel<<<NN, 256, 0, stream>>>(xcur, n1w + (size_t)l * CC, xbf, NN);
    gemm_bf16<0><<<gC, 256, 0, stream>>>(xbf, WqT + (size_t)l * CC * CC, bq + (size_t)l * CC, nullptr, qb, NN, CC, CC);
    gemm_bf16<0><<<gC, 256, 0, stream>>>(xbf, WkT + (size_t)l * CC * CC, bk + (size_t)l * CC, nullptr, kb, NN, CC, CC);
    gemm_bf16<0><<<gC, 256, 0, stream>>>(xbf, WvT + (size_t)l * CC * CC, bv + (size_t)l * CC, nullptr, vb, NN, CC, CC);
    init_mz<<<((NN * HH) + 255) / 256, 256, 0, stream>>>(mb, zb, NN * HH);
    zero_f32<<<((NN * CC) + 255) / 256, 256, 0, stream>>>(agg, NN * CC);
    edge_score<<<(EE + 7) / 8, 256, 0, stream>>>(qb, kb, rel_l, srcv, dstv, etype, score, mb, EE);
    edge_expsum<<<((EE * HH) + 255) / 256, 256, 0, stream>>>(score, mb, zb, dstv, EE);
    edge_agg<<<(EE + 7) / 8, 256, 0, stream>>>(vb, rel_l, srcv, dstv, etype, score, zb, agg, EE);
    cvt_f32_bf16<<<((NN * CC) + 255) / 256, 256, 0, stream>>>(agg, xbf, NN * CC);
    gemm_bf16<1><<<gC, 256, 0, stream>>>(xbf, WoT + (size_t)l * CC * CC, bo + (size_t)l * CC, xcur, xcur, NN, CC, CC);
    // FFN sublayer
    rmsnorm_kernel<<<NN, 256, 0, stream>>>(xcur, n2w + (size_t)l * CC, xbf, NN);
    gemm_bf16<2><<<gF, 256, 0, stream>>>(xbf, W1T + (size_t)l * CC * FFN_, b1 + (size_t)l * FFN_, nullptr, hbuf, NN, FFN_, CC);
    float* outx = (l == LL - 1) ? (float*)d_out : xcur;
    gemm_bf16<1><<<gC, 256, 0, stream>>>(hbuf, W2T + (size_t)l * CC * FFN_, b2 + (size_t)l * CC, xcur, outx, NN, CC, FFN_);
  }
}
